// GemmaAttention_40939628265873
// MI455X (gfx1250) — compile-verified, matrix operand order checked by reading, not run on hardware
//
#include <hip/hip_runtime.h>
#include <hip/hip_bf16.h>

typedef __attribute__((ext_vector_type(16))) _Float16 v16h;
typedef __attribute__((ext_vector_type(8)))  float    v8f;

#define T_     2048
#define NH_    32
#define NKV_   8
#define HD_    128
#define HDIM_  4096
#define QKVO_  6144
#define NEG_INF_ (-1.0e9f)
#define SCALE_   (0.08838834764831845f)   // 128^-0.5

static __device__ __forceinline__ v8f vzero8() {
  v8f r;
#pragma unroll
  for (int i = 0; i < 8; ++i) r[i] = 0.0f;
  return r;
}

// CDNA5 async global->LDS copy: 16B per lane, tracked by ASYNCcnt.
static __device__ __forceinline__ void async_copy_b128(unsigned lds_off, const void* gptr) {
  asm volatile("global_load_async_to_lds_b128 %0, %1, off"
               :: "v"(lds_off), "v"(gptr) : "memory");
}
static __device__ __forceinline__ void wait_async_le4() {
  asm volatile("s_wait_asynccnt 0x4" ::: "memory");
}
static __device__ __forceinline__ void wait_async_0() {
  asm volatile("s_wait_asynccnt 0x0" ::: "memory");
}

// ---------------------------------------------------------------------------
// f32 -> f16 conversion (vectorized x4, grid-stride)
// ---------------------------------------------------------------------------
__global__ void cvt_f32_f16(const float* __restrict__ in, _Float16* __restrict__ out, size_t n4) {
  size_t i = (size_t)blockIdx.x * blockDim.x + threadIdx.x;
  size_t stride = (size_t)gridDim.x * blockDim.x;
  for (; i < n4; i += stride) {
    const float4 v = ((const float4*)in)[i];
    _Float16* o = out + i * 4;
    o[0] = (_Float16)v.x; o[1] = (_Float16)v.y;
    o[2] = (_Float16)v.z; o[3] = (_Float16)v.w;
  }
}

// ---------------------------------------------------------------------------
// GEMM  C[M,N] (f32) = A[M,K] (f16, row-major) * B[N,K]^T (f16, row-major)
// block = 256 threads (8 waves), C tile 128x128, wave subtile 32x64.
// K staged through double-buffered LDS slabs filled by async global->LDS DMA.
// ---------------------------------------------------------------------------
#define TSTR 40   // LDS row stride in halves (80B: keeps 16B alignment, staggers banks)

__global__ __launch_bounds__(256) void gemm_abT(const _Float16* __restrict__ A,
                                                const _Float16* __restrict__ B,
                                                float* __restrict__ C,
                                                int M, int N, int K) {
  __shared__ _Float16 As[2][128 * TSTR];
  __shared__ _Float16 Bs[2][128 * TSTR];

  const int tid  = threadIdx.x;
  const int lane = tid & 31;
  const int wid  = tid >> 5;          // 0..7
  const int wm   = (wid & 3) * 32;    // wave M offset inside tile
  const int wn   = (wid >> 2) * 64;   // wave N offset inside tile
  const int m0   = blockIdx.y * 128;
  const int n0   = blockIdx.x * 128;
  const int lh   = lane & 15;
  const int hi16 = (lane >> 4) * 16;  // K-half select for A/B fragments
  const int hi8  = (lane >> 4) * 8;   // row-half select for C fragments

  v8f acc[2][4];
#pragma unroll
  for (int i = 0; i < 2; ++i)
#pragma unroll
    for (int j = 0; j < 4; ++j) acc[i][j] = vzero8();

  // cooperative-load assignment: thread owns one 32B (16-half) chunk per tile,
  // moved as two async B128 transfers.
  const int r = tid >> 1;            // tile row 0..127
  const int c = (tid & 1) * 16;      // k chunk 0 or 16
  const _Float16* ga = A + (size_t)(m0 + r) * K + c;
  const _Float16* gb = B + (size_t)(n0 + r) * K + c;
  unsigned ldsA[2], ldsB[2];
#pragma unroll
  for (int b = 0; b < 2; ++b) {
    ldsA[b] = (unsigned)(uintptr_t)(&As[b][r * TSTR + c]);
    ldsB[b] = (unsigned)(uintptr_t)(&Bs[b][r * TSTR + c]);
  }

  auto issue_slab = [&](int buf, int k0) {
    async_copy_b128(ldsA[buf],      ga + k0);
    async_copy_b128(ldsA[buf] + 16, ga + k0 + 8);
    async_copy_b128(ldsB[buf],      gb + k0);
    async_copy_b128(ldsB[buf] + 16, gb + k0 + 8);
  };

  const int nk = K >> 5;             // K / 32 slabs
  issue_slab(0, 0);

  for (int i = 0; i < nk; ++i) {
    const int cur = i & 1;
    if (i + 1 < nk) {
      issue_slab((i + 1) & 1, (i + 1) << 5);
      wait_async_le4();              // slab i landed (in-order), slab i+1 in flight
    } else {
      wait_async_0();
    }
    __syncthreads();

    v16h af[2], bf[4];
#pragma unroll
    for (int ii = 0; ii < 2; ++ii)
      af[ii] = *(const v16h*)(&As[cur][(wm + ii * 16 + lh) * TSTR + hi16]);
#pragma unroll
    for (int j = 0; j < 4; ++j)
      bf[j] = *(const v16h*)(&Bs[cur][(wn + j * 16 + lh) * TSTR + hi16]);

#pragma unroll
    for (int ii = 0; ii < 2; ++ii)
#pragma unroll
      for (int j = 0; j < 4; ++j)
        acc[ii][j] = __builtin_amdgcn_wmma_f32_16x16x32_f16(
            false, af[ii], false, bf[j], (short)0, acc[ii][j], false, false);

    __syncthreads();                 // protect buffer from slab i+2 overwrite
  }

#pragma unroll
  for (int i = 0; i < 2; ++i)
#pragma unroll
    for (int j = 0; j < 4; ++j)
#pragma unroll
      for (int v = 0; v < 8; ++v)
        C[(size_t)(m0 + wm + i * 16 + v + hi8) * N + (n0 + wn + j * 16 + lh)] = acc[i][j][v];
}

// ---------------------------------------------------------------------------
// RoPE on q and k heads (from f32 qkv), writing f16 q (t,nh,hd) and k (t,nkv,hd)
// ---------------------------------------------------------------------------
__global__ void rope_qk(const float* __restrict__ qkv, const float* __restrict__ freqs,
                        _Float16* __restrict__ qf, _Float16* __restrict__ kf) {
  const int per_t = (NH_ + NKV_) * 64;
  int idx = blockIdx.x * blockDim.x + threadIdx.x;
  if (idx >= T_ * per_t) return;
  const int t    = idx / per_t;
  const int rem  = idx % per_t;
  const int head = rem / 64;
  const int i    = rem % 64;
  const float cc = freqs[((size_t)t * 64 + i) * 2 + 0];
  const float ss = freqs[((size_t)t * 64 + i) * 2 + 1];
  if (head < NH_) {
    const float* base = qkv + (size_t)t * QKVO_ + head * HD_;
    const float x1 = base[i], x2 = base[i + 64];
    _Float16* o = qf + ((size_t)t * NH_ + head) * HD_;
    o[i]      = (_Float16)(x1 * cc - x2 * ss);
    o[i + 64] = (_Float16)(x1 * ss + x2 * cc);
  } else {
    const int kh = head - NH_;
    const float* base = qkv + (size_t)t * QKVO_ + HDIM_ + kh * HD_;
    const float x1 = base[i], x2 = base[i + 64];
    _Float16* o = kf + ((size_t)t * NKV_ + kh) * HD_;
    o[i]      = (_Float16)(x1 * cc - x2 * ss);
    o[i + 64] = (_Float16)(x1 * ss + x2 * cc);
  }
}

// ---------------------------------------------------------------------------
// V transpose: qkv f32 (t, v-section) -> vT f16 laid out (kv, hd, t)
// ---------------------------------------------------------------------------
__global__ void v_transpose(const float* __restrict__ qkv, _Float16* __restrict__ vT) {
  int idx = blockIdx.x * blockDim.x + threadIdx.x;
  if (idx >= T_ * NKV_ * HD_) return;
  const int t  = idx / (NKV_ * HD_);
  const int kh = (idx / HD_) % NKV_;
  const int d  = idx % HD_;
  vT[((size_t)kh * HD_ + d) * T_ + t] =
      (_Float16)qkv[(size_t)t * QKVO_ + HDIM_ + (size_t)NKV_ * HD_ + kh * HD_ + d];
}

// ---------------------------------------------------------------------------
// Flash attention, one wave per (head, 16-query tile), transposed orientation:
//   S^T = K x Q^T   (both fragments are contiguous row-major loads)
//   out^T = V^T x P^T  (V pre-transposed; P^T built with one shfl_xor(16))
// Softmax runs over the VGPR axis -> per-lane scalars m/l/alpha.
// ---------------------------------------------------------------------------
__global__ __launch_bounds__(32) void flash_attn(const _Float16* __restrict__ qf,
                                                 const _Float16* __restrict__ kf,
                                                 const _Float16* __restrict__ vT,
                                                 _Float16* __restrict__ attn_out) {
  const int h    = blockIdx.x;        // q head
  const int t0   = blockIdx.y * 16;   // query tile base
  const int kh   = h >> 2;            // kv head (G=4)
  const int lane = threadIdx.x;
  const int lh   = lane & 15;
  const int hi   = lane >> 4;

  // Q fragments (B operand): lane=q column, contiguous along hd
  v16h qfrag[4];
  {
    const _Float16* qbase = qf + ((size_t)(t0 + lh) * NH_ + h) * HD_;
#pragma unroll
    for (int kb = 0; kb < 4; ++kb)
      qfrag[kb] = *(const v16h*)(qbase + kb * 32 + hi * 16);
  }

  v8f acc[8];
#pragma unroll
  for (int n = 0; n < 8; ++n) acc[n] = vzero8();
  float m_run = -1.0e30f, l_run = 0.0f;
  const int q_pos = t0 + lh;

  for (int sb = 0; sb < t0 + 16; sb += 32) {
    // ---- S^T scores for two 16-key blocks
    v8f s0 = vzero8(), s1 = vzero8();
#pragma unroll
    for (int blk = 0; blk < 2; ++blk) {
      const int s16 = sb + blk * 16;
      const _Float16* kbase = kf + ((size_t)(s16 + lh) * NKV_ + kh) * HD_;
      v8f sc = vzero8();
#pragma unroll
      for (int kb = 0; kb < 4; ++kb) {
        v16h kfrag = *(const v16h*)(kbase + kb * 32 + hi * 16);
        sc = __builtin_amdgcn_wmma_f32_16x16x32_f16(
            false, kfrag, false, qfrag[kb], (short)0, sc, false, false);
      }
      if (blk == 0) s0 = sc; else s1 = sc;
    }

    // ---- scale + causal mask (row = key s, col = query q=lane)
#pragma unroll
    for (int v = 0; v < 8; ++v) {
      const int sl = sb + v + hi * 8;
      s0[v] = (sl      <= q_pos) ? s0[v] * SCALE_ : NEG_INF_;
      s1[v] = (sl + 16 <= q_pos) ? s1[v] * SCALE_ : NEG_INF_;
    }

    // ---- online softmax over s (VGPR axis + partner half-wave)
    float mx = NEG_INF_;
#pragma unroll
    for (int v = 0; v < 8; ++v) { mx = fmaxf(mx, s0[v]); mx = fmaxf(mx, s1[v]); }
    mx = fmaxf(mx, __shfl_xor(mx, 16, 32));
    const float m_new = fmaxf(m_run, mx);
    const float alpha = __expf(m_run - m_new);
    float psum = 0.0f;
    v8f p0, p1;
#pragma unroll
    for (int v = 0; v < 8; ++v) {
      const float e0 = __expf(s0[v] - m_new);
      const float e1 = __expf(s1[v] - m_new);
      p0[v] = e0; p1[v] = e1;
      psum += e0 + e1;
    }
    psum += __shfl_xor(psum, 16, 32);
    l_run = l_run * alpha + psum;
    m_run = m_new;

    // ---- build P^T B-fragment (k = s_local 0..31, n = q)
    v16h pb;
#pragma unroll
    for (int v = 0; v < 8; ++v) {
      const float o0 = __shfl_xor(p0[v], 16, 32);
      const float o1 = __shfl_xor(p1[v], 16, 32);
      const _Float16 lo = hi ? (_Float16)o1 : (_Float16)p0[v];
      const _Float16 hv = hi ? (_Float16)p1[v] : (_Float16)o0;
      pb[v]     = lo;
      pb[v + 8] = hv;
    }

    // ---- out^T += V^T x P^T, with alpha rescale folded into C
    const _Float16* vbase = vT + (size_t)kh * HD_ * T_ + sb + hi * 16;
#pragma unroll
    for (int n = 0; n < 8; ++n) {
#pragma unroll
      for (int v = 0; v < 8; ++v) acc[n][v] *= alpha;
      v16h vfrag = *(const v16h*)(vbase + (size_t)(n * 16 + lh) * T_);
      acc[n] = __builtin_amdgcn_wmma_f32_16x16x32_f16(
          false, vfrag, false, pb, (short)0, acc[n], false, false);
    }
  }

  // ---- normalize and write (transpose back on store)
  const float inv_l = 1.0f / l_run;
  _Float16* obase = attn_out + (size_t)(t0 + lh) * HDIM_ + h * HD_;
#pragma unroll
  for (int n = 0; n < 8; ++n)
#pragma unroll
    for (int v = 0; v < 8; ++v)
      obase[n * 16 + v + hi * 8] = (_Float16)(acc[n][v] * inv_l);
}

// ---------------------------------------------------------------------------
// Launch
// ---------------------------------------------------------------------------
extern "C" void kernel_launch(void* const* d_in, const int* in_sizes, int n_in,
                              void* d_out, int out_size, void* d_ws, size_t ws_size,
                              hipStream_t stream) {
  const float* hs     = (const float*)d_in[0];   // (1,T,H)
  const float* freqs  = (const float*)d_in[1];   // (T,64,2)
  const float* qkv_w  = (const float*)d_in[6];   // (6144,4096)
  const float* o_w    = (const float*)d_in[7];   // (4096,4096)
  float* out = (float*)d_out;

  char* ws = (char*)d_ws;
  size_t off = 0;
  _Float16* hs_h   = (_Float16*)(ws + off); off += (size_t)T_ * HDIM_ * 2;          // 16 MB
  _Float16* qkvw_h = (_Float16*)(ws + off); off += (size_t)QKVO_ * HDIM_ * 2;       // 50 MB
  _Float16* ow_h   = (_Float16*)(ws + off); off += (size_t)HDIM_ * HDIM_ * 2;       // 34 MB
  float*    qkv    = (float*)(ws + off);    off += (size_t)T_ * QKVO_ * 4;          // 50 MB
  _Float16* q_h    = (_Float16*)(ws + off); off += (size_t)T_ * NH_ * HD_ * 2;      // 16 MB
  _Float16* k_h    = (_Float16*)(ws + off); off += (size_t)T_ * NKV_ * HD_ * 2;     //  4 MB
  _Float16* vT_h   = (_Float16*)(ws + off); off += (size_t)T_ * NKV_ * HD_ * 2;     //  4 MB
  _Float16* attn_h = (_Float16*)(ws + off); off += (size_t)T_ * HDIM_ * 2;          // 16 MB

  // 1) f32 -> f16 staging
  cvt_f32_f16<<<4096, 256, 0, stream>>>(hs,    hs_h,   (size_t)T_ * HDIM_ / 4);
  cvt_f32_f16<<<8192, 256, 0, stream>>>(qkv_w, qkvw_h, (size_t)QKVO_ * HDIM_ / 4);
  cvt_f32_f16<<<8192, 256, 0, stream>>>(o_w,   ow_h,   (size_t)HDIM_ * HDIM_ / 4);

  // 2) QKV projection: (T x H) * (O x H)^T
  gemm_abT<<<dim3(QKVO_ / 128, T_ / 128), 256, 0, stream>>>(hs_h, qkvw_h, qkv, T_, QKVO_, HDIM_);

  // 3) RoPE(q,k) + V transpose
  {
    const int n = T_ * (NH_ + NKV_) * 64;
    rope_qk<<<(n + 255) / 256, 256, 0, stream>>>(qkv, freqs, q_h, k_h);
    const int nv = T_ * NKV_ * HD_;
    v_transpose<<<(nv + 255) / 256, 256, 0, stream>>>(qkv, vT_h);
  }

  // 4) flash attention
  flash_attn<<<dim3(NH_, T_ / 16), 32, 0, stream>>>(q_h, k_h, vT_h, attn_h);

  // 5) output projection: (T x O) * (H x O)^T
  gemm_abT<<<dim3(HDIM_ / 128, T_ / 128), 256, 0, stream>>>(attn_h, ow_h, out, T_, HDIM_, HDIM_);
}
